// SelfAttentionLayer_33543694581939
// MI455X (gfx1250) — matrix operand order repaired and checked by reading, not compiled, not run on hardware
//
#include <hip/hip_runtime.h>
#include <hip/hip_bf16.h>

#define BB 8
#define SS 2048
#define DD 1024

typedef __attribute__((ext_vector_type(16))) __bf16 v16bf;
typedef __attribute__((ext_vector_type(8)))  float  v8f;

union FragU {
  v16bf v;
  uint4 q[2];
  unsigned short h[16];
};

static __device__ __forceinline__ v8f wmma_bf16(v16bf a, v16bf b, v8f c) {
  // D = A(16x32 bf16) * B(32x16 bf16) + C(16x16 f32)
  return __builtin_amdgcn_wmma_f32_16x16x32_bf16(
      /*neg_a=*/false, a, /*neg_b=*/false, b,
      /*c_mod=*/(short)0, c, /*reuse_a=*/false, /*reuse_b=*/false);
}

static __device__ __forceinline__ unsigned int packbf2(float a, float b) {
  unsigned int ua = __float_as_uint(a), ub = __float_as_uint(b);
  ua = (ua + 0x7FFFu + ((ua >> 16) & 1u)) >> 16;  // round-to-nearest-even
  ub = (ub + 0x7FFFu + ((ub >> 16) & 1u)) >> 16;
  return (ua & 0xFFFFu) | (ub << 16);
}
static __device__ __forceinline__ unsigned short f2bf(float a) {
  unsigned int u = __float_as_uint(a);
  return (unsigned short)((u + 0x7FFFu + ((u >> 16) & 1u)) >> 16);
}

// ---------------------------------------------------------------------------
// Kernel 0: one-shot fp32 -> bf16 conversion (bandwidth bound; each element
// converted exactly once instead of 64x/1024x inside the GEMM k-loop).
// 8 elements per thread, 16B-vectorized in and out. n must be a multiple of 8.
// ---------------------------------------------------------------------------
__global__ __launch_bounds__(256)
void cvt_f32_to_bf16_kernel(const float* __restrict__ src,
                            unsigned short* __restrict__ dst,
                            long long n) {
  const long long stride = (long long)gridDim.x * blockDim.x * 8;
  for (long long i = ((long long)blockIdx.x * blockDim.x + threadIdx.x) * 8;
       i < n; i += stride) {
    float4 f0 = *(const float4*)(src + i);
    float4 f1 = *(const float4*)(src + i + 4);
    uint4 o = make_uint4(packbf2(f0.x, f0.y), packbf2(f0.z, f0.w),
                         packbf2(f1.x, f1.y), packbf2(f1.z, f1.w));
    *(uint4*)(dst + i) = o;
  }
}

// ---------------------------------------------------------------------------
// Kernel 1: Out[b,s,e] = sum_d Xbf[b,s,d] * Wbf[e,d] + bias[e] (bf16 WMMA,
// fp32 accumulate). Hot loop is pure global_load_b128 + v_wmma.
// transposed==0 -> Out row-major [B,S,D]; transposed==1 -> Out = [B,D,S] (Vt).
// Block = 256 threads (8 waves). WG tile: 16 rows (M) x 128 cols (N);
// each wave owns one 16x16 N-tile, K=1024 consumed in 32-wide WMMA steps.
// ---------------------------------------------------------------------------
__global__ __launch_bounds__(256)
void proj_bf16_kernel(const unsigned short* __restrict__ Xbf,
                      const unsigned short* __restrict__ Wbf,
                      const float* __restrict__ bias,
                      unsigned short* __restrict__ Out,
                      int transposed) {
  const int tid  = threadIdx.x;
  const int wave = tid >> 5;
  const int lane = tid & 31;
  const int half = lane >> 4;
  const int ln   = lane & 15;

  const int blk      = blockIdx.x;
  const int ncol_blk = blk & 7;          // D/128 = 8
  const int mtile    = (blk >> 3) & 127; // S/16  = 128
  const int b        = blk >> 10;

  const int mbase = mtile * 16;
  const int nbase = ncol_blk * 128 + wave * 16;

  // A-frag rows: lane ln holds row (mbase+ln); K interleaved by 8 per half.
  const unsigned short* xrow = Xbf + ((size_t)b * SS + mbase + ln) * DD;
  // B-frag cols: lane ln holds W row (nbase+ln); K interleaved by 8 per half.
  const unsigned short* wrow = Wbf + (size_t)(nbase + ln) * DD;

  v8f c = {};
#pragma unroll 4
  for (int kd = 0; kd < DD; kd += 32) {
    FragU a, bf;
    a.q[0]  = *(const uint4*)(xrow + kd + half * 8);
    a.q[1]  = *(const uint4*)(xrow + kd + 16 + half * 8);
    bf.q[0] = *(const uint4*)(wrow + kd + half * 8);
    bf.q[1] = *(const uint4*)(wrow + kd + 16 + half * 8);
    c = wmma_bf16(a.v, bf.v, c);
  }

  const float bv = bias[nbase + ln];
  if (!transposed) {
#pragma unroll
    for (int i = 0; i < 8; ++i) {
      const int m = mbase + half * 8 + i;
      Out[((size_t)b * SS + m) * DD + nbase + ln] = f2bf(c[i] + bv);
    }
  } else {
    // Vt[b, n, m]: lane's 8 rows are contiguous -> single 16B store.
    unsigned short* dst =
        Out + ((size_t)b * DD + nbase + ln) * SS + mbase + half * 8;
    uint4 o = make_uint4(packbf2(c[0] + bv, c[1] + bv),
                         packbf2(c[2] + bv, c[3] + bv),
                         packbf2(c[4] + bv, c[5] + bv),
                         packbf2(c[6] + bv, c[7] + bv));
    *(uint4*)dst = o;
  }
}

// ---------------------------------------------------------------------------
// Kernel 2: per-WG = (batch b, 16 query rows).
// Phase 1: scores 16x2048 via WMMA, wave w owns keys [w*256, w*256+256).
// Phase 2: scale + causal/pad mask on accumulators, softmax with shfl_xor +
//          cross-wave LDS reductions; writes fp32 attn + bf16 attn to LDS.
// Phase 3: seq_emds = attn_bf16(LDS) x Vt via WMMA, wave w owns d-cols
//          [w*128, w*128+128).
// ---------------------------------------------------------------------------
#define ATTN_PITCH 2056                       // 2048 + 8 (bank stagger)
#define LDS_ATTN_BYTES (16 * ATTN_PITCH * 2)  // 65792
#define LDS_RED0_OFF   LDS_ATTN_BYTES
#define LDS_RED1_OFF   (LDS_RED0_OFF + 8 * 16 * 4)
#define LDS_TOTAL      (LDS_RED1_OFF + 8 * 16 * 4)  // 66816 bytes

__global__ __launch_bounds__(256)
void attn_kernel(const unsigned short* __restrict__ Qbf,
                 const unsigned short* __restrict__ Kbf,
                 const unsigned short* __restrict__ Vt,
                 const unsigned char* __restrict__ pad,
                 float* __restrict__ attn_out,
                 float* __restrict__ emb_out) {
  extern __shared__ unsigned char smem[];
  unsigned short* lds_attn = (unsigned short*)(smem);
  float* red0 = (float*)(smem + LDS_RED0_OFF);
  float* red1 = (float*)(smem + LDS_RED1_OFF);

  const int tid  = threadIdx.x;
  const int wave = tid >> 5;
  const int lane = tid & 31;
  const int half = lane >> 4;
  const int ln   = lane & 15;

  const int b     = blockIdx.x >> 7;   // S/16 = 128 q-tiles per batch
  const int qtile = blockIdx.x & 127;
  const int qbase = qtile * 16;

  const v8f zero = {};

  // ---- Phase 1: scores = Q . K^T over d=1024 ----
  v8f acc[16];
#pragma unroll
  for (int t = 0; t < 16; ++t) acc[t] = zero;

  const unsigned short* qrow = Qbf + ((size_t)b * SS + qbase + ln) * DD;
  const size_t kbb = (size_t)b * SS;
  for (int kd = 0; kd < DD; kd += 32) {
    FragU a;
    a.q[0] = *(const uint4*)(qrow + kd + half * 8);
    a.q[1] = *(const uint4*)(qrow + kd + 16 + half * 8);
#pragma unroll
    for (int t = 0; t < 16; ++t) {
      const int key = wave * 256 + t * 16 + ln;
      const unsigned short* krow = Kbf + (kbb + key) * DD;
      FragU bf;
      bf.q[0] = *(const uint4*)(krow + kd + half * 8);
      bf.q[1] = *(const uint4*)(krow + kd + 16 + half * 8);
      acc[t] = wmma_bf16(a.v, bf.v, acc[t]);
    }
  }

  // ---- Phase 2: scale, mask, softmax ----
  const float scale = 0.03125f;  // 1/sqrt(1024)
#pragma unroll
  for (int t = 0; t < 16; ++t) {
    const int key = wave * 256 + t * 16 + ln;
    const bool kvalid = pad[kbb + key] != 0;
#pragma unroll
    for (int i = 0; i < 8; ++i) {
      const int q = qbase + half * 8 + i;
      const bool valid = kvalid && (key <= q);  // causal & pad (combined mask)
      acc[t][i] = valid ? acc[t][i] * scale : -1.0e30f;
    }
  }

  // row max: per-lane over 16 tiles, then shfl_xor across the 16-lane half.
  float gmax[8];
  {
    float rmax[8];
#pragma unroll
    for (int i = 0; i < 8; ++i) {
      float m = acc[0][i];
#pragma unroll
      for (int t = 1; t < 16; ++t) m = fmaxf(m, acc[t][i]);
#pragma unroll
      for (int d = 1; d < 16; d <<= 1) m = fmaxf(m, __shfl_xor(m, d, 32));
      rmax[i] = m;
    }
    if (ln == 0) {
#pragma unroll
      for (int i = 0; i < 8; ++i) red0[wave * 16 + half * 8 + i] = rmax[i];
    }
    __syncthreads();
#pragma unroll
    for (int i = 0; i < 8; ++i) {
      const int row = half * 8 + i;
      float m = red0[row];
#pragma unroll
      for (int wv = 1; wv < 8; ++wv) m = fmaxf(m, red0[wv * 16 + row]);
      gmax[i] = m;
    }
  }

  // exp + row sum
  float rsum[8];
#pragma unroll
  for (int i = 0; i < 8; ++i) rsum[i] = 0.0f;
#pragma unroll
  for (int t = 0; t < 16; ++t) {
#pragma unroll
    for (int i = 0; i < 8; ++i) {
      const float e = __expf(acc[t][i] - gmax[i]);
      acc[t][i] = e;
      rsum[i] += e;
    }
  }
#pragma unroll
  for (int i = 0; i < 8; ++i) {
#pragma unroll
    for (int d = 1; d < 16; d <<= 1) rsum[i] += __shfl_xor(rsum[i], d, 32);
  }
  if (ln == 0) {
#pragma unroll
    for (int i = 0; i < 8; ++i) red1[wave * 16 + half * 8 + i] = rsum[i];
  }
  __syncthreads();

  float fac[8];
#pragma unroll
  for (int i = 0; i < 8; ++i) {
    const int row = half * 8 + i;
    float s = 0.0f;
#pragma unroll
    for (int wv = 0; wv < 8; ++wv) s += red1[wv * 16 + row];
    const bool rowvalid = gmax[i] > -1.0e29f;            // row_any
    const bool padq = pad[kbb + qbase + row] != 0;       // query-axis pad mult
    fac[i] = (rowvalid && padq) ? (1.0f / s) : 0.0f;
  }

  // write fp32 attn + stage bf16 attn into LDS
#pragma unroll
  for (int t = 0; t < 16; ++t) {
    const int keyl = wave * 256 + t * 16 + ln;
#pragma unroll
    for (int i = 0; i < 8; ++i) {
      const int row = half * 8 + i;
      const float av = acc[t][i] * fac[i];
      attn_out[((size_t)b * SS + qbase + row) * SS + keyl] = av;
      lds_attn[row * ATTN_PITCH + keyl] = f2bf(av);
    }
  }
  __syncthreads();

  // ---- Phase 3: seq_emds = attn x V  (A from LDS, B from Vt) ----
  v8f acc2[8];
#pragma unroll
  for (int t = 0; t < 8; ++t) acc2[t] = zero;

  const unsigned short* arow = lds_attn + ln * ATTN_PITCH;
  for (int kk = 0; kk < SS; kk += 32) {
    FragU a;
    a.q[0] = *(const uint4*)(arow + kk + half * 8);
    a.q[1] = *(const uint4*)(arow + kk + 16 + half * 8);
#pragma unroll
    for (int t = 0; t < 8; ++t) {
      const int dcol = wave * 128 + t * 16 + ln;
      const unsigned short* vrow = Vt + ((size_t)b * DD + dcol) * SS;
      FragU bf;
      bf.q[0] = *(const uint4*)(vrow + kk + half * 8);
      bf.q[1] = *(const uint4*)(vrow + kk + 16 + half * 8);
      acc2[t] = wmma_bf16(a.v, bf.v, acc2[t]);
    }
  }
#pragma unroll
  for (int t = 0; t < 8; ++t) {
#pragma unroll
    for (int i = 0; i < 8; ++i) {
      emb_out[((size_t)b * SS + qbase + half * 8 + i) * DD +
              wave * 128 + t * 16 + ln] = acc2[t][i];
    }
  }
}

// ---------------------------------------------------------------------------
extern "C" void kernel_launch(void* const* d_in, const int* in_sizes, int n_in,
                              void* d_out, int out_size, void* d_ws, size_t ws_size,
                              hipStream_t stream) {
  (void)in_sizes; (void)n_in; (void)out_size; (void)ws_size;

  const float* X[3]  = {(const float*)d_in[0], (const float*)d_in[1],
                        (const float*)d_in[2]};               // query,key,value
  const unsigned char* pad = (const unsigned char*)d_in[3];
  // d_in[4] = causal_masks: recomputed in-kernel as (key <= query)
  const float* W[3]  = {(const float*)d_in[5], (const float*)d_in[7],
                        (const float*)d_in[9]};               // Wq,Wk,Wv
  const float* bias[3] = {(const float*)d_in[6], (const float*)d_in[8],
                          (const float*)d_in[10]};            // bq,bk,bv

  const size_t elems = (size_t)BB * SS * DD;  // 16.78M
  const size_t welems = (size_t)DD * DD;      // 1.05M
  unsigned short* Qbf = (unsigned short*)d_ws;
  unsigned short* Kbf = Qbf + elems;
  unsigned short* Vt  = Kbf + elems;   // [B, D, S]
  unsigned short* Xbf = Vt + elems;    // staging: current input in bf16
  unsigned short* Wbf = Xbf + elems;   // staging: current weight in bf16
  unsigned short* outs[3] = {Qbf, Kbf, Vt};

  float* emb  = (float*)d_out;          // seq_emds: B*S*D
  float* attn = emb + elems;            // attn:     B*S*S

  hipFuncSetAttribute((const void*)attn_kernel,
                      hipFuncAttributeMaxDynamicSharedMemorySize, LDS_TOTAL);

  const dim3 blk(256);
  const dim3 pgrid(BB * (SS / 16) * (DD / 128));  // 8192 WGs

  for (int m = 0; m < 3; ++m) {
    cvt_f32_to_bf16_kernel<<<dim3(2048), blk, 0, stream>>>(
        X[m], Xbf, (long long)elems);
    cvt_f32_to_bf16_kernel<<<dim3(256), blk, 0, stream>>>(
        W[m], Wbf, (long long)welems);
    proj_bf16_kernel<<<pgrid, blk, 0, stream>>>(
        Xbf, Wbf, bias[m], outs[m], (m == 2) ? 1 : 0);
  }

  const dim3 agrid(BB * (SS / 16));  // 1024 WGs
  attn_kernel<<<agrid, blk, LDS_TOTAL, stream>>>(Qbf, Kbf, Vt, pad, attn, emb);
}
